// VectorQuantizer_90640989815347
// MI455X (gfx1250) — matrix operand order repaired and checked by reading, not compiled, not run on hardware
//
#include <hip/hip_runtime.h>
#include <hip/hip_bf16.h>

typedef __attribute__((ext_vector_type(16))) _Float16 v16h;
typedef __attribute__((ext_vector_type(8)))  float    v8f;

#define VQ_K 1024
#define VQ_D 64

// ---------------------------------------------------------------------------
// Kernel 0: codebook prep. emb f32 -> f16 (for WMMA B operand), per-code ||e||^2,
// zero the count/mse accumulators (harness poisons ws with 0xAA).
// ---------------------------------------------------------------------------
__global__ __launch_bounds__(256) void vq_prep_kernel(
    const float* __restrict__ emb, _Float16* __restrict__ embh,
    float* __restrict__ e2, float* __restrict__ counts, float* __restrict__ mse)
{
  int tid = blockIdx.x * 256 + threadIdx.x;   // 65536 threads total
  embh[tid] = (_Float16)emb[tid];
  if (tid < VQ_K) {
    const float* r = emb + tid * VQ_D;
    float s = 0.f;
    #pragma unroll 8
    for (int i = 0; i < VQ_D; ++i) s += r[i] * r[i];
    e2[tid] = s;
    counts[tid] = 0.f;
  }
  if (tid == 0) *mse = 0.f;
}

// ---------------------------------------------------------------------------
// Kernel 1 (fused): WMMA distance matmul + argmin + gather + loss/count accum.
// Block = 256 threads = 8 waves; each wave owns 16 rows of x and scans all
// K=1024 codes held in LDS (f16). Distances scored as ||e||^2 - 2*x.e
// (||x||^2 is constant per row, irrelevant for argmin).
// ---------------------------------------------------------------------------
__global__ __launch_bounds__(256) void vq_main_kernel(
    const float* __restrict__ x, const float* __restrict__ emb32,
    const _Float16* __restrict__ embh, const float* __restrict__ e2g,
    float* __restrict__ out, float* __restrict__ idxf,
    float* __restrict__ counts, float* __restrict__ mse)
{
  extern __shared__ char smem[];
  _Float16* semb = (_Float16*)smem;                        // 128 KB codebook (f16)
  float*    se2  = (float*)(smem + VQ_K * VQ_D * 2);       // 4 KB  ||e||^2
  int*      sidx = (int*)(smem + VQ_K * VQ_D * 2 + VQ_K * 4); // 8 waves x 16 rows

  const int t = threadIdx.x;

  // Cooperative codebook load: 131072 B = 8192 uint4 across 256 threads.
  {
    const uint4* src = (const uint4*)embh;
    uint4* dst = (uint4*)semb;
    #pragma unroll
    for (int i = 0; i < 32; ++i) dst[t + i * 256] = src[t + i * 256];
    for (int i = t; i < VQ_K; i += 256) se2[i] = e2g[i];
  }
  __syncthreads();

  const int wave = t >> 5;
  const int lane = t & 31;
  const int m    = lane & 15;   // A: row within tile / C: code column
  const int hf   = lane >> 4;   // lane-half select
  const int rowtile = blockIdx.x * 8 + wave;   // 16-row tile index

  // ---- Build WMMA A operands (16 rows x K=64, f32 -> f16) -----------------
  // A 16x32 layout: lane half h holds K = h*16 .. h*16+15 (contiguous);
  // A[1] covers K=32..63.
  const float* xrow = x + (size_t)(rowtile * 16 + m) * VQ_D;
  const int kb = hf * 16;
  v16h A[2];
  #pragma unroll
  for (int s = 0; s < 4; ++s) {
    float4 lo = *(const float4*)(xrow + (s >> 1) * 32 + (s & 1) * 8 + kb);
    float4 hi = *(const float4*)(xrow + (s >> 1) * 32 + (s & 1) * 8 + kb + 4);
    const int base = (s & 1) * 8;
    v16h& Av = A[s >> 1];
    Av[base + 0] = (_Float16)lo.x; Av[base + 1] = (_Float16)lo.y;
    Av[base + 2] = (_Float16)lo.z; Av[base + 3] = (_Float16)lo.w;
    Av[base + 4] = (_Float16)hi.x; Av[base + 5] = (_Float16)hi.y;
    Av[base + 6] = (_Float16)hi.z; Av[base + 7] = (_Float16)hi.w;
  }

  float bestv[8];
  int   besti[8];
  #pragma unroll
  for (int v = 0; v < 8; ++v) { bestv[v] = __builtin_inff(); besti[v] = 0; }

  // ---- Scan all 64 code tiles (16 codes each) via WMMA --------------------
  for (int ct = 0; ct < VQ_K / 16; ++ct) {
    const int code = ct * 16 + m;   // B operand: lane's code column
    // B 32x16 layout: lane half h holds K = h*16 .. h*16+15 (contiguous f16).
    const _Float16* erow = semb + code * VQ_D + hf * 16;
    v16h B0 = *(const v16h*)(erow);        // K =  0..31 slice for this half
    v16h B1 = *(const v16h*)(erow + 32);   // K = 32..63 slice
    v8f c = {};
    c = __builtin_amdgcn_wmma_f32_16x16x32_f16(false, A[0], false, B0,
                                               (short)0, c, false, false);
    c = __builtin_amdgcn_wmma_f32_16x16x32_f16(false, A[1], false, B1,
                                               (short)0, c, false, false);
    const float ee = se2[code];
    // C layout: VGPR v / lane-half hf -> row hf*8+v, column m (== this code).
    #pragma unroll
    for (int v = 0; v < 8; ++v) {
      float d = ee - 2.0f * c[v];
      if (d < bestv[v]) { bestv[v] = d; besti[v] = code; }
    }
  }

  // ---- Argmin reduction across the 16 code-columns (lanes within a half) --
  #pragma unroll
  for (int v = 0; v < 8; ++v) {
    float bv = bestv[v];
    int   bi = besti[v];
    #pragma unroll
    for (int off = 8; off >= 1; off >>= 1) {
      float ov = __shfl_xor(bv, off, 32);
      int   oi = __shfl_xor(bi, off, 32);
      if (ov < bv || (ov == bv && oi < bi)) { bv = ov; bi = oi; }
    }
    if (m == 0) sidx[wave * 16 + hf * 8 + v] = bi;   // row hf*8+v winner
  }
  __syncthreads();

  // ---- Fused gather + straight-through output + loss/count accumulation ---
  float lsq = 0.f;
  #pragma unroll 4
  for (int r = 0; r < 16; ++r) {
    const int code = sidx[wave * 16 + r];
    const size_t grow = (size_t)(rowtile * 16 + r);
    const float* er   = emb32 + (size_t)code * VQ_D;  // full-precision codebook (L2-hot)
    const float* xr   = x + grow * VQ_D;
    float*       orow = out + grow * VQ_D;
    #pragma unroll
    for (int j = 0; j < 2; ++j) {
      const int col = lane + j * 32;
      float q  = er[col];
      float xv = xr[col];
      orow[col] = q;                       // quantized_st == quantized numerically
      float df = q - xv;
      lsq += df * df;
    }
    if (lane == 0) {
      atomicAdd(counts + code, 1.0f);
      idxf[grow] = (float)code;
    }
  }
  #pragma unroll
  for (int off = 16; off >= 1; off >>= 1) lsq += __shfl_xor(lsq, off, 32);
  if (lane == 0) atomicAdd(mse, lsq);
}

// ---------------------------------------------------------------------------
// Kernel 2: scalars. loss = (1 + 0.25) * mean((q-x)^2); perplexity from counts.
// ---------------------------------------------------------------------------
__global__ __launch_bounds__(256) void vq_final_kernel(
    const float* __restrict__ counts, const float* __restrict__ mse,
    float* __restrict__ loss_out, float* __restrict__ perp_out, int N)
{
  __shared__ float red[256];
  const float invN = 1.0f / (float)N;
  float s = 0.f;
  for (int i = threadIdx.x; i < VQ_K; i += 256) {
    float p = counts[i] * invN;
    s += p * logf(p + 1e-10f);
  }
  red[threadIdx.x] = s;
  __syncthreads();
  for (int k = 128; k > 0; k >>= 1) {
    if (threadIdx.x < k) red[threadIdx.x] += red[threadIdx.x + k];
    __syncthreads();
  }
  if (threadIdx.x == 0) {
    *perp_out = expf(-red[0]);
    *loss_out = (*mse / (float)(N * VQ_D)) * (1.0f + 0.25f);
  }
}

// ---------------------------------------------------------------------------
extern "C" void kernel_launch(void* const* d_in, const int* in_sizes, int n_in,
                              void* d_out, int out_size, void* d_ws, size_t ws_size,
                              hipStream_t stream) {
  (void)n_in; (void)out_size; (void)ws_size;
  const float* x   = (const float*)d_in[0];   // (32,64,64,64) f32
  const float* emb = (const float*)d_in[1];   // (1024,64) f32
  const int n_elem = in_sizes[0];             // 8388608
  const int N = n_elem / VQ_D;                // 131072 rows

  // Workspace layout
  char* ws = (char*)d_ws;
  _Float16* embh  = (_Float16*)ws;                            // 128 KB f16 codebook
  float*    e2    = (float*)(ws + VQ_K * VQ_D * 2);           // 4 KB
  float*    cnts  = (float*)(ws + VQ_K * VQ_D * 2 + VQ_K * 4);// 4 KB
  float*    mse   = (float*)(ws + VQ_K * VQ_D * 2 + VQ_K * 8);// 4 B

  // Output layout: [quantized_st | loss | perplexity | indices(as float)]
  float* out  = (float*)d_out;
  float* loss = out + (size_t)n_elem;
  float* perp = loss + 1;
  float* idxf = perp + 1;

  vq_prep_kernel<<<(VQ_K * VQ_D) / 256, 256, 0, stream>>>(emb, embh, e2, cnts, mse);

  const size_t lds = (size_t)VQ_K * VQ_D * 2 + VQ_K * 4 + 8 * 16 * 4; // ~132.5 KB
  hipFuncSetAttribute((const void*)vq_main_kernel,
                      hipFuncAttributeMaxDynamicSharedMemorySize, (int)lds);
  const int blocks = N / (16 * 8);   // 1024 blocks, 8 waves x 16 rows each
  vq_main_kernel<<<blocks, 256, lds, stream>>>(x, emb, embh, e2, out, idxf, cnts, mse);

  vq_final_kernel<<<1, 256, 0, stream>>>(cnts, mse, loss, perp, N);
}
